// Block_11166914969721
// MI455X (gfx1250) — matrix operand order repaired and checked by reading, not compiled, not run on hardware
//
#include <hip/hip_runtime.h>
#include <hip/hip_bf16.h>
#include <cmath>

// ---------------------------------------------------------------------------
// CDNA5 (gfx1250) implementation: all GEMMs + flash attention on
// v_wmma_f32_16x16x32_f16 (wave32), fp32 accumulate, f16 operands.
// V is staged TRANSPOSED (dim-major) so every WMMA operand fragment is
// loaded with contiguous global_load_b128 pairs (no scalar gathers).
// ---------------------------------------------------------------------------

typedef __attribute__((ext_vector_type(16))) _Float16 v16h;
typedef __attribute__((ext_vector_type(8)))  float    v8f;

#define BB     2
#define C_     384
#define NPTS   4096      // 16*16*16
#define HD     48        // head dim
#define HDP    64        // padded head dim (k=32 WMMA steps divide evenly)
#define N1PTS  512       // 8*8*8
#define HID    1536

union F16x16 { uint4 u[2]; v16h h; };

__device__ __forceinline__ v8f wmma_f16(v16h a, v16h b, v8f c) {
    // D = A(16x32) * B(32x16) + C, fp32 accumulate
    return __builtin_amdgcn_wmma_f32_16x16x32_f16(false, a, false, b, (short)0, c,
                                                  false, false);
}

// A fragment from row-major [*, ldK] f16: lane lo = row, elems cover
// k = 16*hi + e  (contiguous, matching the B fragments)
__device__ __forceinline__ v16h frag_a_rowmajor(const _Float16* base, int ldK) {
    int lane = threadIdx.x & 31, lo = lane & 15, hi = lane >> 4;
    const _Float16* p = base + (size_t)lo * ldK + 16 * hi;
    F16x16 f;
    f.u[0] = *(const uint4*)(p);
    f.u[1] = *(const uint4*)(p + 8);
    return f.h;
}

// B fragment for  X @ W^T : B[k][n] = W[n][k];  lane lo = column n,
// elems are k = 16*hi + e  -> 16 contiguous halfs of one weight row.
__device__ __forceinline__ v16h frag_b_wT(const _Float16* base, int ldK) {
    int lane = threadIdx.x & 31, lo = lane & 15, hi = lane >> 4;
    const _Float16* p = base + (size_t)lo * ldK + 16 * hi;
    F16x16 f;
    f.u[0] = *(const uint4*)(p);
    f.u[1] = *(const uint4*)(p + 8);
    return f.h;
}

__device__ __forceinline__ float gelu_exact(float x) {
    return 0.5f * x * (1.0f + erff(x * 0.70710678118654752f));
}

// ---------------------------------------------------------------------------
// Generic GEMM  out[M,N] = A[M,K] @ W[N,K]^T   (wave tile 16x64, 4 WMMA/iter)
// ---------------------------------------------------------------------------
constexpr int EPI_F32 = 0, EPI_BIAS_F32 = 1, EPI_F16 = 2,
              EPI_BIAS_GELU_F16 = 3, EPI_BIAS_ADD_F32 = 4;

template <int EPI>
__global__ __launch_bounds__(256)
void gemm_xwT(const _Float16* __restrict__ A, const _Float16* __restrict__ W,
              const float* __restrict__ bias, const float* __restrict__ addres,
              float* __restrict__ outF, _Float16* __restrict__ outH,
              int M, int N, int K) {
    int wave = (blockIdx.x * blockDim.x + threadIdx.x) >> 5;
    int tilesN = N >> 6, tilesM = M >> 4;
    if (wave >= tilesM * tilesN) return;
    int mt = wave / tilesN, nt = wave % tilesN;
    int m0 = mt << 4, n0 = nt << 6;

    v8f acc0 = {}, acc1 = {}, acc2 = {}, acc3 = {};
    for (int k0 = 0; k0 < K; k0 += 32) {
        v16h a = frag_a_rowmajor(A + (size_t)m0 * K + k0, K);
        acc0 = wmma_f16(a, frag_b_wT(W + (size_t)(n0)      * K + k0, K), acc0);
        acc1 = wmma_f16(a, frag_b_wT(W + (size_t)(n0 + 16) * K + k0, K), acc1);
        acc2 = wmma_f16(a, frag_b_wT(W + (size_t)(n0 + 32) * K + k0, K), acc2);
        acc3 = wmma_f16(a, frag_b_wT(W + (size_t)(n0 + 48) * K + k0, K), acc3);
    }
    int lane = threadIdx.x & 31, lo = lane & 15, hi = lane >> 4;
    v8f* accs[4] = {&acc0, &acc1, &acc2, &acc3};
#pragma unroll
    for (int j = 0; j < 4; ++j) {
        int col = n0 + j * 16 + lo;
#pragma unroll
        for (int r = 0; r < 8; ++r) {
            int row = m0 + r + 8 * hi;
            size_t idx = (size_t)row * N + col;
            float v = (*accs[j])[r];
            if (EPI == EPI_F32)              outF[idx] = v;
            else if (EPI == EPI_BIAS_F32)    outF[idx] = v + bias[col];
            else if (EPI == EPI_F16)         outH[idx] = (_Float16)v;
            else if (EPI == EPI_BIAS_GELU_F16)
                outH[idx] = (_Float16)gelu_exact(v + bias[col]);
            else  /* EPI_BIAS_ADD_F32 */     outF[idx] = v + bias[col] + addres[idx];
        }
    }
}

// ---------------------------------------------------------------------------
// Flash attention (streaming softmax), one wave per (b, head, 16-query tile).
//   S^T = K * Q^T  -> softmax over element dim -> O^T += V^T * P^T
// V supplied pre-transposed: Vt[(b*4+h)*48*NK + dim*NK + key]
// ---------------------------------------------------------------------------
__global__ __launch_bounds__(128)
void flash_attn(const _Float16* __restrict__ Qp,  // [B*8, NPTS, HDP]
                const _Float16* __restrict__ Kp,  // [B*4, NK, HDP]
                const _Float16* __restrict__ Vt,  // [B*4, 48, NK]  (dim-major!)
                _Float16* __restrict__ att16,     // [B*NPTS, 384]
                int NK, int headBase, int chBase, float scale) {
    int wave = (blockIdx.x * blockDim.x + threadIdx.x) >> 5;
    int lane = threadIdx.x & 31, lo = lane & 15, hi = lane >> 4;
    int qt = wave & 255;            // NPTS/16 = 256 query tiles
    int head = (wave >> 8) & 3;
    int b = wave >> 10;
    if (b >= BB) return;
    int q0 = qt << 4;

    const _Float16* Qb = Qp + ((size_t)(b * 8 + headBase + head) * NPTS + q0) * HDP;
    const _Float16* Kb = Kp + (size_t)(b * 4 + head) * NK * HDP;
    const _Float16* Vb = Vt + (size_t)(b * 4 + head) * HD * NK;

    // Q^T B-fragments (dims 0..31, 32..63); lane lo = query column
    v16h bq0, bq1;
    {
        const _Float16* p = Qb + (size_t)lo * HDP + 16 * hi;
        F16x16 f; f.u[0] = *(const uint4*)p;        f.u[1] = *(const uint4*)(p + 8);  bq0 = f.h;
        F16x16 g; g.u[0] = *(const uint4*)(p + 32); g.u[1] = *(const uint4*)(p + 40); bq1 = g.h;
    }

    float mrun = -1e30f, lrun = 0.0f;
    v8f o0 = {}, o1 = {}, o2 = {};

    for (int kc = 0; kc < NK; kc += 32) {
        // S^T tiles: tile0 = keys kc..kc+15, tile1 = kc+16..kc+31
        v8f s0 = {}, s1 = {};
        s0 = wmma_f16(frag_a_rowmajor(Kb + (size_t)kc * HDP,       HDP), bq0, s0);
        s0 = wmma_f16(frag_a_rowmajor(Kb + (size_t)kc * HDP + 32,  HDP), bq1, s0);
        s1 = wmma_f16(frag_a_rowmajor(Kb + (size_t)(kc + 16) * HDP,      HDP), bq0, s1);
        s1 = wmma_f16(frag_a_rowmajor(Kb + (size_t)(kc + 16) * HDP + 32, HDP), bq1, s1);

        // online softmax: rows (keys) are register elements, query = lane pair
        float mloc = -1e30f;
#pragma unroll
        for (int r = 0; r < 8; ++r) {
            s0[r] *= scale; s1[r] *= scale;
            mloc = fmaxf(mloc, fmaxf(s0[r], s1[r]));
        }
        mloc = fmaxf(mloc, __shfl_xor(mloc, 16, 32));
        float mnew = fmaxf(mrun, mloc);
        float alpha = __expf(mrun - mnew);
        float p0[8], p1[8], ls = 0.0f;
#pragma unroll
        for (int r = 0; r < 8; ++r) {
            p0[r] = __expf(s0[r] - mnew);
            p1[r] = __expf(s1[r] - mnew);
            ls += p0[r] + p1[r];
        }
        ls += __shfl_xor(ls, 16, 32);
        lrun = lrun * alpha + ls;
        mrun = mnew;
#pragma unroll
        for (int r = 0; r < 8; ++r) { o0[r] *= alpha; o1[r] *= alpha; o2[r] *= alpha; }

        // repack P^T into a B fragment (keys in element dim, query = lane lo)
        v16h pb = {};
#pragma unroll
        for (int r = 0; r < 8; ++r) {
            float t0 = __shfl_xor(p0[r], 16, 32);  // partner tile0
            float t1 = __shfl_xor(p1[r], 16, 32);  // partner tile1
            pb[r]     = (_Float16)(hi ? t1 : p0[r]);   // keys 16*hi + r
            pb[r + 8] = (_Float16)(hi ? p1[r] : t0);   // keys 16*hi + 8 + r
        }

        // O^T += V^T * P^T  (3 dim-tiles cover 48 dims); V is dim-major, so
        // each A fragment is two contiguous b128 loads per lane.
        o0 = wmma_f16(frag_a_rowmajor(Vb + (size_t)0  * NK + kc, NK), pb, o0);
        o1 = wmma_f16(frag_a_rowmajor(Vb + (size_t)16 * NK + kc, NK), pb, o1);
        o2 = wmma_f16(frag_a_rowmajor(Vb + (size_t)32 * NK + kc, NK), pb, o2);
    }

    float inv = 1.0f / lrun;
    int ch0 = chBase + head * HD;
    size_t row = (size_t)(b * NPTS + q0 + lo) * C_;
#pragma unroll
    for (int r = 0; r < 8; ++r) {
        att16[row + ch0 + 0  + r + 8 * hi] = (_Float16)(o0[r] * inv);
        att16[row + ch0 + 16 + r + 8 * hi] = (_Float16)(o1[r] * inv);
        att16[row + ch0 + 32 + r + 8 * hi] = (_Float16)(o2[r] * inv);
    }
}

// ---------------------------------------------------------------------------
// LayerNorm (wave per row), optional residual-add input, GELU, f32 residual out
// ---------------------------------------------------------------------------
template <bool ADD, bool GELU, bool STORE_RES>
__global__ __launch_bounds__(256)
void layernorm_k(const float* __restrict__ X, const float* __restrict__ Y,
                 const float* __restrict__ w, const float* __restrict__ b,
                 float* __restrict__ res, _Float16* __restrict__ out16, int rows) {
    int wave = (blockIdx.x * blockDim.x + threadIdx.x) >> 5;
    int lane = threadIdx.x & 31;
    if (wave >= rows) return;
    const float* xr = X + (size_t)wave * C_;
    const float* yr = Y + (size_t)wave * C_;
    float s1 = 0.f, s2 = 0.f;
    for (int c = lane; c < C_; c += 32) {
        float v = xr[c]; if (ADD) v += yr[c];
        s1 += v; s2 += v * v;
    }
#pragma unroll
    for (int off = 16; off; off >>= 1) {
        s1 += __shfl_xor(s1, off, 32);
        s2 += __shfl_xor(s2, off, 32);
    }
    float mean = s1 * (1.0f / C_);
    float var  = s2 * (1.0f / C_) - mean * mean;
    float rstd = rsqrtf(var + 1e-5f);
    for (int c = lane; c < C_; c += 32) {
        float v = xr[c]; if (ADD) v += yr[c];
        if (STORE_RES) res[(size_t)wave * C_ + c] = v;
        float t = (v - mean) * rstd * w[c] + b[c];
        if (GELU) t = gelu_exact(t);
        out16[(size_t)wave * C_ + c] = (_Float16)t;
    }
}

// ---------------------------------------------------------------------------
// Small elementwise kernels
// ---------------------------------------------------------------------------
__global__ __launch_bounds__(256)
void cvt_f16(const float* __restrict__ s, _Float16* __restrict__ d, int n) {
    int g = blockIdx.x * blockDim.x + threadIdx.x;
    if (g < n) d[g] = (_Float16)s[g];
}

// im2col for sr1 (k=2, s=2):  col[(b*512+n1)*3072 + i*8 + kz*4+ky*2+kx]
__global__ __launch_bounds__(256)
void im2col_sr1(const _Float16* __restrict__ xa, _Float16* __restrict__ col, int total) {
    int g = blockIdx.x * blockDim.x + threadIdx.x;
    if (g >= total) return;
    int cidx = g % 3072, row = g / 3072;
    int n1 = row % N1PTS, b = row / N1PTS;
    int i = cidx >> 3, kk = cidx & 7;
    int kz = kk >> 2, ky = (kk >> 1) & 1, kx = kk & 1;
    int z = n1 >> 6, y = (n1 >> 3) & 7, x = n1 & 7;
    int n = ((2 * z + kz) * 16 + (2 * y + ky)) * 16 + (2 * x + kx);
    col[g] = xa[((size_t)b * NPTS + n) * C_ + i];
}

// q f32 [B*N,384] -> padded f16 [B*8, N, 64]
__global__ __launch_bounds__(256)
void scatter_q(const float* __restrict__ qf, _Float16* __restrict__ qpad, int total) {
    int g = blockIdx.x * blockDim.x + threadIdx.x;
    if (g >= total) return;
    int d = g & 63, rest = g >> 6;
    int n = rest & (NPTS - 1), bh = rest >> 12;
    int h = bh & 7, b = bh >> 3;
    qpad[g] = (d < HD) ? (_Float16)qf[((size_t)b * NPTS + n) * C_ + h * HD + d]
                       : (_Float16)0.0f;
}

// kv f32 [B*Nk,384] -> padded K f16 [B*4, Nk, 64] and V f16 [B*4, Nk, 64]
__global__ __launch_bounds__(256)
void scatter_kv(const float* __restrict__ kvf, _Float16* __restrict__ kpad,
                _Float16* __restrict__ vpad, int Nk, int total) {
    int g = blockIdx.x * blockDim.x + threadIdx.x;
    if (g >= total) return;
    int d = g & 63, rest = g >> 6;
    int n = rest % Nk, bh = rest / Nk;
    int h = bh & 3, b = bh >> 2;
    _Float16 kv = (_Float16)0.0f, vv = (_Float16)0.0f;
    if (d < HD) {
        const float* rowp = kvf + ((size_t)b * Nk + n) * C_;
        kv = (_Float16)rowp[h * HD + d];
        vv = (_Float16)rowp[192 + h * HD + d];
    }
    kpad[g] = kv; vpad[g] = vv;
}

// depthwise 3x3x3 SAME conv + bias + residual on V; reads key-major [B*4,S^3,64],
// writes TRANSPOSED dim-major [B*4, 48, S^3] for the flash-attention V^T GEMM.
__global__ __launch_bounds__(256)
void dwconv_add_t(const _Float16* __restrict__ vin, _Float16* __restrict__ voutT,
                  const float* __restrict__ w, const float* __restrict__ bias,
                  int S, int total) {
    int g = blockIdx.x * blockDim.x + threadIdx.x;
    if (g >= total) return;
    int d = g & 63, rest = g >> 6;
    if (d >= HD) return;                      // pad lanes: nothing to emit
    int NkS = S * S * S;
    int n = rest % NkS, bh = rest / NkS;
    int h = bh & 3;
    int c2 = h * HD + d;
    int z = n / (S * S), y = (n / S) % S, x = n % S;
    float acc = bias[c2];
    for (int kz = 0; kz < 3; ++kz) {
        int zz = z + kz - 1; if (zz < 0 || zz >= S) continue;
        for (int ky = 0; ky < 3; ++ky) {
            int yy = y + ky - 1; if (yy < 0 || yy >= S) continue;
            for (int kx = 0; kx < 3; ++kx) {
                int xx = x + kx - 1; if (xx < 0 || xx >= S) continue;
                int nn = (zz * S + yy) * S + xx;
                acc += (float)vin[((size_t)bh * NkS + nn) * HDP + d]
                       * w[c2 * 27 + (kz * 3 + ky) * 3 + kx];
            }
        }
    }
    voutT[((size_t)bh * HD + d) * NkS + n] =
        (_Float16)((float)vin[((size_t)bh * NkS + n) * HDP + d] + acc);
}

// ---------------------------------------------------------------------------
// Host-side launcher
// ---------------------------------------------------------------------------
static inline size_t alignup(size_t x) { return (x + 255) & ~(size_t)255; }

extern "C" void kernel_launch(void* const* d_in, const int* in_sizes, int n_in,
                              void* d_out, int out_size, void* d_ws, size_t ws_size,
                              hipStream_t stream) {
    const float* x      = (const float*)d_in[0];
    const float* ln1_w  = (const float*)d_in[1];
    const float* ln1_b  = (const float*)d_in[2];
    const float* q_w    = (const float*)d_in[3];
    const float* sr1_w  = (const float*)d_in[4];
    const float* sr1_b  = (const float*)d_in[5];
    const float* n1_w   = (const float*)d_in[6];
    const float* n1_b   = (const float*)d_in[7];
    const float* sr2_w  = (const float*)d_in[8];
    const float* sr2_b  = (const float*)d_in[9];
    const float* n2_w   = (const float*)d_in[10];
    const float* n2_b   = (const float*)d_in[11];
    const float* kv1_w  = (const float*)d_in[12];
    const float* kv2_w  = (const float*)d_in[13];
    const float* lc1_w  = (const float*)d_in[14];
    const float* lc1_b  = (const float*)d_in[15];
    const float* lc2_w  = (const float*)d_in[16];
    const float* lc2_b  = (const float*)d_in[17];
    const float* proj_w = (const float*)d_in[18];
    const float* proj_b = (const float*)d_in[19];
    const float* ln2_w  = (const float*)d_in[20];
    const float* ln2_b  = (const float*)d_in[21];
    const float* fc1_w  = (const float*)d_in[22];
    const float* fc1_b  = (const float*)d_in[23];
    const float* fc2_w  = (const float*)d_in[24];
    const float* fc2_b  = (const float*)d_in[25];
    float* out = (float*)d_out;

    const int ROWS = BB * NPTS;          // 8192
    const int R1   = BB * N1PTS;         // 1024
    const float scale = 0.14433756729740643f;  // 48^-0.5

    // ---- workspace carve-out (f16 activations, f32 temps) ----
    char* p = (char*)d_ws; size_t off = 0;
    auto take = [&](size_t bytes) { char* r = p + off; off = alignup(off + bytes); return r; };

    _Float16* xa16  = (_Float16*)take((size_t)ROWS * C_ * 2);
    _Float16* wq16  = (_Float16*)take((size_t)C_ * C_ * 2);
    _Float16* wsr1  = (_Float16*)take((size_t)C_ * 3072 * 2);
    _Float16* wsr2  = (_Float16*)take((size_t)C_ * C_ * 2);
    _Float16* wkv1  = (_Float16*)take((size_t)C_ * C_ * 2);
    _Float16* wkv2  = (_Float16*)take((size_t)C_ * C_ * 2);
    _Float16* wproj = (_Float16*)take((size_t)C_ * C_ * 2);
    _Float16* wfc1  = (_Float16*)take((size_t)HID * C_ * 2);
    _Float16* wfc2  = (_Float16*)take((size_t)C_ * HID * 2);
    float*    qf32  = (float*)take((size_t)ROWS * C_ * 4);     // reused as attout
    _Float16* qpad  = (_Float16*)take((size_t)BB * 8 * NPTS * HDP * 2);
    _Float16* col1  = (_Float16*)take((size_t)R1 * 3072 * 2);
    float*    y1f   = (float*)take((size_t)R1 * C_ * 4);
    _Float16* x1_16 = (_Float16*)take((size_t)R1 * C_ * 2);
    float*    y2f   = (float*)take((size_t)ROWS * C_ * 4);     // reused as kv2f
    _Float16* x2_16 = (_Float16*)take((size_t)ROWS * C_ * 2);
    float*    kv1f  = (float*)take((size_t)R1 * C_ * 4);
    _Float16* k1p   = (_Float16*)take((size_t)BB * 4 * N1PTS * HDP * 2);
    _Float16* v1p   = (_Float16*)take((size_t)BB * 4 * N1PTS * HDP * 2);
    _Float16* v1t   = (_Float16*)take((size_t)BB * 4 * HD * N1PTS * 2);  // transposed
    _Float16* k2p   = (_Float16*)take((size_t)BB * 4 * NPTS * HDP * 2);
    _Float16* v2p   = (_Float16*)take((size_t)BB * 4 * NPTS * HDP * 2);
    _Float16* v2t   = (_Float16*)take((size_t)BB * 4 * HD * NPTS * 2);   // transposed
    _Float16* att16 = (_Float16*)take((size_t)ROWS * C_ * 2);
    float*    xres  = (float*)take((size_t)ROWS * C_ * 4);
    _Float16* xm16  = (_Float16*)take((size_t)ROWS * C_ * 2);
    _Float16* hid16 = (_Float16*)take((size_t)ROWS * HID * 2);
    float*    attoutf = qf32;   // alias (qf32 dead after scatter_q)
    float*    kv2f    = y2f;    // alias (y2f dead after n2 LayerNorm)

    auto ceil_div = [](int a, int b) { return (a + b - 1) / b; };

    // ---- weight conversions to f16 ----
    auto cvt = [&](const float* s, _Float16* d, int n) {
        cvt_f16<<<ceil_div(n, 256), 256, 0, stream>>>(s, d, n);
    };
    cvt(q_w,    wq16,  C_ * C_);
    cvt(sr1_w,  wsr1,  C_ * 3072);
    cvt(sr2_w,  wsr2,  C_ * C_);
    cvt(kv1_w,  wkv1,  C_ * C_);
    cvt(kv2_w,  wkv2,  C_ * C_);
    cvt(proj_w, wproj, C_ * C_);
    cvt(fc1_w,  wfc1,  HID * C_);
    cvt(fc2_w,  wfc2,  C_ * HID);

    // ---- LN1 -> xa16 ----
    layernorm_k<false, false, false><<<ROWS / 8, 256, 0, stream>>>(
        x, x, ln1_w, ln1_b, nullptr, xa16, ROWS);

    // ---- q = xa @ q_w^T, scatter into padded [B*8,N,64] ----
    gemm_xwT<EPI_F32><<<(ROWS / 16) * (C_ / 64) / 8, 256, 0, stream>>>(
        xa16, wq16, nullptr, nullptr, qf32, nullptr, ROWS, C_, C_);
    {
        int tot = BB * 8 * NPTS * HDP;
        scatter_q<<<ceil_div(tot, 256), 256, 0, stream>>>(qf32, qpad, tot);
    }

    // ---- branch 1: im2col + conv-GEMM + LN + GELU ----
    {
        int tot = R1 * 3072;
        im2col_sr1<<<ceil_div(tot, 256), 256, 0, stream>>>(xa16, col1, tot);
    }
    gemm_xwT<EPI_BIAS_F32><<<(R1 / 16) * (C_ / 64) / 8, 256, 0, stream>>>(
        col1, wsr1, sr1_b, nullptr, y1f, nullptr, R1, C_, 3072);
    layernorm_k<false, true, false><<<R1 / 8, 256, 0, stream>>>(
        y1f, y1f, n1_w, n1_b, nullptr, x1_16, R1);

    // ---- branch 2: 1x1x1 conv (GEMM) + LN + GELU ----
    gemm_xwT<EPI_BIAS_F32><<<(ROWS / 16) * (C_ / 64) / 8, 256, 0, stream>>>(
        xa16, wsr2, sr2_b, nullptr, y2f, nullptr, ROWS, C_, C_);
    layernorm_k<false, true, false><<<ROWS / 8, 256, 0, stream>>>(
        y2f, y2f, n2_w, n2_b, nullptr, x2_16, ROWS);

    // ---- kv projections + scatter into padded K/V ----
    gemm_xwT<EPI_F32><<<(R1 / 16) * (C_ / 64) / 8, 256, 0, stream>>>(
        x1_16, wkv1, nullptr, nullptr, kv1f, nullptr, R1, C_, C_);
    {
        int tot = BB * 4 * N1PTS * HDP;
        scatter_kv<<<ceil_div(tot, 256), 256, 0, stream>>>(kv1f, k1p, v1p, N1PTS, tot);
    }
    gemm_xwT<EPI_F32><<<(ROWS / 16) * (C_ / 64) / 8, 256, 0, stream>>>(
        x2_16, wkv2, nullptr, nullptr, kv2f, nullptr, ROWS, C_, C_);
    {
        int tot = BB * 4 * NPTS * HDP;
        scatter_kv<<<ceil_div(tot, 256), 256, 0, stream>>>(kv2f, k2p, v2p, NPTS, tot);
    }

    // ---- local-context depthwise 3x3x3 conv on V (+residual), emit V^T ----
    {
        int tot1 = BB * 4 * N1PTS * HDP;
        dwconv_add_t<<<ceil_div(tot1, 256), 256, 0, stream>>>(v1p, v1t, lc1_w, lc1_b, 8, tot1);
        int tot2 = BB * 4 * NPTS * HDP;
        dwconv_add_t<<<ceil_div(tot2, 256), 256, 0, stream>>>(v2p, v2t, lc2_w, lc2_b, 16, tot2);
    }

    // ---- flash attention (both branches write into att16 channel halves) ----
    {
        int waves = BB * 4 * (NPTS / 16);            // 2048
        flash_attn<<<waves / 4, 128, 0, stream>>>(qpad, k1p, v1t, att16,
                                                  N1PTS, 0, 0, scale);
        flash_attn<<<waves / 4, 128, 0, stream>>>(qpad, k2p, v2t, att16,
                                                  NPTS, 4, 192, scale);
    }

    // ---- proj + residual + LN2 ----
    gemm_xwT<EPI_BIAS_F32><<<(ROWS / 16) * (C_ / 64) / 8, 256, 0, stream>>>(
        att16, wproj, proj_b, nullptr, attoutf, nullptr, ROWS, C_, C_);
    layernorm_k<true, false, true><<<ROWS / 8, 256, 0, stream>>>(
        x, attoutf, ln2_w, ln2_b, xres, xm16, ROWS);

    // ---- MLP: fc1 (+GELU) then fc2 (+bias +residual) -> d_out ----
    gemm_xwT<EPI_BIAS_GELU_F16><<<(ROWS / 16) * (HID / 64) / 8, 256, 0, stream>>>(
        xm16, wfc1, fc1_b, nullptr, nullptr, hid16, ROWS, HID, C_);
    gemm_xwT<EPI_BIAS_ADD_F32><<<(ROWS / 16) * (C_ / 64) / 8, 256, 0, stream>>>(
        hid16, wfc2, fc2_b, xres, out, nullptr, ROWS, C_, HID);

    (void)in_sizes; (void)n_in; (void)out_size; (void)ws_size;
}
